// Rank2Block_35416300323172
// MI455X (gfx1250) — matrix operand order repaired and checked by reading, not compiled, not run on hardware
//
#include <hip/hip_runtime.h>
#include <stdint.h>

// ---------------------------------------------------------------------------
// Rank2Block on MI455X (gfx1250, wave32).
//   Y = x_edge @ W1    -> bf16x3 split WMMA (v_wmma_f32_16x16x32_bf16)
//   e[e,k] = sum_m silu(op_k*Y[e,m]+b1[m])*W2[m] + b2   (6 unique of 9 comps)
//   node mean -> graph mean via f32 atomics (counts clamped to 1)
// X tile staged to LDS via TENSOR_LOAD_TO_LDS (TDM, 6-arg builtin form).
// ---------------------------------------------------------------------------

typedef __attribute__((ext_vector_type(16))) __bf16    v16bf;
typedef __attribute__((ext_vector_type(8)))  float     v8f;
typedef __attribute__((ext_vector_type(8)))  uint32_t  v8u;
typedef __attribute__((ext_vector_type(4)))  uint32_t  v4u;
typedef __attribute__((ext_vector_type(8)))  int32_t   v8i;
typedef __attribute__((ext_vector_type(4)))  int32_t   v4i;

#define EMB     256
#define TILE_E  64      // edges per workgroup
#define XP      264     // LDS pitch (bf16) for staged X hi/lo
#define YP      260     // LDS pitch (f32) for Y tile

#if __has_builtin(__builtin_amdgcn_tensor_load_to_lds) && \
    __has_builtin(__builtin_amdgcn_s_wait_tensorcnt)
#define USE_TDM 1
#else
#define USE_TDM 0
#endif

static __device__ __forceinline__ uint16_t f32_to_bf16_rne(float f) {
  uint32_t u = __float_as_uint(f);
  uint32_t r = u + 0x7FFFu + ((u >> 16) & 1u);   // round-to-nearest-even
  return (uint16_t)(r >> 16);
}
static __device__ __forceinline__ float bf16_to_f32(uint16_t h) {
  return __uint_as_float(((uint32_t)h) << 16);
}

// B frag (32x16 bf16): lane holds one column; 8 contiguous uints (16 bf16)
static __device__ __forceinline__ v16bf load_bfrag(const uint16_t* p) {
  const uint4* q = reinterpret_cast<const uint4*>(p);
  uint4 x0 = q[0], x1 = q[1];
  v8u u = {x0.x, x0.y, x0.z, x0.w, x1.x, x1.y, x1.z, x1.w};
  return __builtin_bit_cast(v16bf, u);
}
// A frag (16x32 bf16): half 0 -> K{0..15}, half 1 -> K{16..31}
static __device__ __forceinline__ v16bf load_afrag(const uint16_t* p, int half) {
  const uint4* q = reinterpret_cast<const uint4*>(p);
  uint4 x0 = q[2 * half];
  uint4 x1 = q[2 * half + 1];
  v8u u = {x0.x, x0.y, x0.z, x0.w, x1.x, x1.y, x1.z, x1.w};
  return __builtin_bit_cast(v16bf, u);
}

#define WMMA_BF16(A, B, C) \
  __builtin_amdgcn_wmma_f32_16x16x32_bf16(false, (A), false, (B), (short)0, (C), false, false)

// --- W1 -> transposed (N-major) bf16 hi/lo split; L2-resident (256KB) -------
__global__ void prep_w_split(const float* __restrict__ W1,
                             uint16_t* __restrict__ Whi,
                             uint16_t* __restrict__ Wlo) {
  int idx = blockIdx.x * blockDim.x + threadIdx.x;
  if (idx >= EMB * EMB) return;
  int n = idx / EMB, k = idx % EMB;          // Wt[n][k] = W1[k][n]
  float w = W1[k * EMB + n];
  uint16_t hb = f32_to_bf16_rne(w);
  Whi[idx] = hb;
  Wlo[idx] = f32_to_bf16_rne(w - bf16_to_f32(hb));
}

__global__ void zero_f32(float* __restrict__ p, int n) {
  int i = blockIdx.x * blockDim.x + threadIdx.x;
  for (; i < n; i += gridDim.x * blockDim.x) p[i] = 0.0f;
}

// --- main fused kernel ------------------------------------------------------
__global__ __launch_bounds__(256, 1)
void edge_mlp_kernel(const float* __restrict__ xv,        // (E,3)
                     const float* __restrict__ xedge,     // (E,256)
                     const uint16_t* __restrict__ Whi,
                     const uint16_t* __restrict__ Wlo,
                     const float* __restrict__ b1,
                     const float* __restrict__ W2,
                     const float* __restrict__ b2,
                     const int* __restrict__ edge_index,
                     float* __restrict__ node_sum,        // (N,9)
                     float* __restrict__ node_cnt,        // (N,)
                     int E) {
  __shared__ uint16_t sXhi[TILE_E * XP];
  __shared__ uint16_t sXlo[TILE_E * XP];
  __shared__ float    sY  [TILE_E * YP];   // also fp32 staging for the TDM copy
  __shared__ float    sB1 [EMB];
  __shared__ float    sW2 [EMB];

  const int t     = threadIdx.x;        // 0..255, 8 waves
  const int wave  = t >> 5;
  const int lane  = t & 31;
  const int half  = lane >> 4;
  const int r16   = lane & 15;
  const int ebase = blockIdx.x * TILE_E;
  const int rows  = (E - ebase < TILE_E) ? (E - ebase) : TILE_E;
  (void)rows;

  // Warm L2 for this wave's B columns (global_prefetch_b8).
  __builtin_prefetch(Whi + (size_t)((wave * 2) * 16 + r16) * EMB, 0, 1);
  __builtin_prefetch(Wlo + (size_t)((wave * 2) * 16 + r16) * EMB, 0, 1);

  sB1[t] = b1[t];
  sW2[t] = W2[t];

#if USE_TDM
  // ---- DMA the 64x256 fp32 X tile into LDS with the Tensor Data Mover -----
  // D# per ISA ch.8: 2-D tile, data_size=4B, tile 256 x rows, row stride 256.
  // Staging buffer aliases sY (dead until after the GEMM).
  if (t < 32) {                               // one wave issues; EXEC ignored
    const uint64_t gaddr = (uint64_t)(uintptr_t)(xedge + (size_t)ebase * EMB);
    const uint32_t laddr = (uint32_t)(uintptr_t)&sY[0];  // LDS byte offset
    v4u g0;
    g0[0] = 1u;                                          // count=1, user D#
    g0[1] = laddr;                                       // lds_addr
    g0[2] = (uint32_t)(gaddr & 0xFFFFFFFFu);             // global_addr lo
    g0[3] = (uint32_t)((gaddr >> 32) & 0x01FFFFFFu) | (2u << 30); // hi | type=2
    v8i g1;
    g1[0] = (int)(2u << 16);            // workgroup_mask=0, data_size=4B
    g1[1] = (int)((uint32_t)EMB << 16); // tensor_dim0[15:0] @ bits 63:48
    g1[2] = (int)((uint32_t)rows << 16);// tensor_dim0 hi=0, tensor_dim1 lo
    g1[3] = (int)((uint32_t)EMB << 16); // tensor_dim1 hi=0, tile_dim0=256
    g1[4] = (int)rows;                  // tile_dim1=rows, tile_dim2=0
    g1[5] = (int)EMB;                   // tensor_dim0_stride lo32 = 256
    g1[6] = 0;                          // stride hi, tensor_dim1_stride lo
    g1[7] = 0;
    v4i gz4 = {0, 0, 0, 0};             // D# groups 2/3 unused for 2-D tile
    v8i gz8 = {0, 0, 0, 0, 0, 0, 0, 0}; // extra group (6-arg builtin form)
    __builtin_amdgcn_tensor_load_to_lds(g0, g1, gz4, gz4, gz8, 0);
    __builtin_amdgcn_s_wait_tensorcnt(0);
  }
  __syncthreads();
  // ---- convert staged fp32 -> bf16 hi/lo split (packed b64 stores) --------
  #pragma unroll
  for (int i = 0; i < 16; ++i) {
    int f4  = t + 256 * i;              // float4 slot 0..4095
    int row = f4 >> 6;
    int col = (f4 & 63) << 2;
    float4 x = reinterpret_cast<const float4*>(&sY[0])[f4];
#else
  #pragma unroll
  for (int i = 0; i < 16; ++i) {
    int f4  = t + 256 * i;
    int row = f4 >> 6;
    int col = (f4 & 63) << 2;
    float4 x = make_float4(0.f, 0.f, 0.f, 0.f);
    if (ebase + row < E)
      x = reinterpret_cast<const float4*>(xedge + (size_t)(ebase + row) * EMB)[col >> 2];
#endif
    float xs[4] = {x.x, x.y, x.z, x.w};
    uint32_t hp[2], lp[2];
    #pragma unroll
    for (int c = 0; c < 2; ++c) {
      uint16_t h0 = f32_to_bf16_rne(xs[2 * c + 0]);
      uint16_t h1 = f32_to_bf16_rne(xs[2 * c + 1]);
      uint16_t l0 = f32_to_bf16_rne(xs[2 * c + 0] - bf16_to_f32(h0));
      uint16_t l1 = f32_to_bf16_rne(xs[2 * c + 1] - bf16_to_f32(h1));
      hp[c] = (uint32_t)h0 | ((uint32_t)h1 << 16);
      lp[c] = (uint32_t)l0 | ((uint32_t)l1 << 16);
    }
    int o = row * XP + col;             // 8B aligned (col%4==0, XP even)
    *reinterpret_cast<uint2*>(&sXhi[o]) = make_uint2(hp[0], hp[1]);
    *reinterpret_cast<uint2*>(&sXlo[o]) = make_uint2(lp[0], lp[1]);
  }
  __syncthreads();

  // ---- GEMM: wave -> 2 column tiles x 4 row blocks; A loaded once/(kc,rb) --
  v8f acc[2][4];
  {
    v8f z = {0.f, 0.f, 0.f, 0.f, 0.f, 0.f, 0.f, 0.f};
    #pragma unroll
    for (int i = 0; i < 2; ++i)
      #pragma unroll
      for (int j = 0; j < 4; ++j) acc[i][j] = z;
  }
  const uint16_t* wh0 = Whi + (size_t)((wave * 2 + 0) * 16 + r16) * EMB;
  const uint16_t* wl0 = Wlo + (size_t)((wave * 2 + 0) * 16 + r16) * EMB;
  const uint16_t* wh1 = Whi + (size_t)((wave * 2 + 1) * 16 + r16) * EMB;
  const uint16_t* wl1 = Wlo + (size_t)((wave * 2 + 1) * 16 + r16) * EMB;

  #pragma unroll 2                       // bounded unroll: no scratch spills
  for (int kc = 0; kc < 8; ++kc) {
    const int ko = kc * 32 + half * 16;
    v16bf Bh0 = load_bfrag(wh0 + ko);
    v16bf Bl0 = load_bfrag(wl0 + ko);
    v16bf Bh1 = load_bfrag(wh1 + ko);
    v16bf Bl1 = load_bfrag(wl1 + ko);
    #pragma unroll
    for (int rb = 0; rb < 4; ++rb) {
      const int ao = (rb * 16 + r16) * XP + kc * 32;
      v16bf Ah = load_afrag(&sXhi[ao], half);
      v16bf Al = load_afrag(&sXlo[ao], half);
      // bf16x3 split: hi*hi + hi*lo + lo*hi  (lo*lo ~2^-16, dropped)
      acc[0][rb] = WMMA_BF16(Ah, Bh0, acc[0][rb]);
      acc[0][rb] = WMMA_BF16(Ah, Bl0, acc[0][rb]);
      acc[0][rb] = WMMA_BF16(Al, Bh0, acc[0][rb]);
      acc[1][rb] = WMMA_BF16(Ah, Bh1, acc[1][rb]);
      acc[1][rb] = WMMA_BF16(Ah, Bl1, acc[1][rb]);
      acc[1][rb] = WMMA_BF16(Al, Bh1, acc[1][rb]);
    }
  }
  __syncthreads();                       // staging reads done -> sY reusable

  // ---- spill Y (fp32) to LDS: C layout VGPR v -> row v (+8 for hi lanes) --
  #pragma unroll
  for (int ct2 = 0; ct2 < 2; ++ct2) {
    const int col = (wave * 2 + ct2) * 16 + r16;
    #pragma unroll
    for (int rb = 0; rb < 4; ++rb) {
      #pragma unroll
      for (int v = 0; v < 8; ++v)
        sY[(rb * 16 + half * 8 + v) * YP + col] = acc[ct2][rb][v];
    }
  }
  __syncthreads();

  // ---- per-edge SiLU dot: 4 threads/edge, 6 unique outer-product comps ----
  const int e  = t >> 2;
  const int s  = t & 3;
  const int eg = ebase + e;
  float v0 = 0.f, v1 = 0.f, v2 = 0.f;
  if (eg < E) {
    v0 = xv[(size_t)eg * 3 + 0];
    v1 = xv[(size_t)eg * 3 + 1];
    v2 = xv[(size_t)eg * 3 + 2];
  }
  const float op0 = v0 * v0, op1 = v0 * v1, op2 = v0 * v2;
  const float op3 = v1 * v1, op4 = v1 * v2, op5 = v2 * v2;
  float a0 = 0.f, a1 = 0.f, a2 = 0.f, a3 = 0.f, a4 = 0.f, a5 = 0.f;
  const float* yrow = &sY[e * YP];

  #pragma unroll 4
  for (int j = 0; j < EMB / 4; ++j) {
    const int m = s + 4 * j;            // stride-4: conflict-free banks
    const float ym = yrow[m];
    const float bb = sB1[m];
    const float w2 = sW2[m];
#define SILU_ACC(OPV, ACC)                                            \
    {                                                                 \
      float z  = fmaf((OPV), ym, bb);                                 \
      float si = z * __builtin_amdgcn_rcpf(1.0f + __expf(-z));        \
      (ACC) = fmaf(si, w2, (ACC));                                    \
    }
    SILU_ACC(op0, a0) SILU_ACC(op1, a1) SILU_ACC(op2, a2)
    SILU_ACC(op3, a3) SILU_ACC(op4, a4) SILU_ACC(op5, a5)
#undef SILU_ACC
  }
  #pragma unroll
  for (int off = 1; off < 4; off <<= 1) {
    a0 += __shfl_xor(a0, off, 32);
    a1 += __shfl_xor(a1, off, 32);
    a2 += __shfl_xor(a2, off, 32);
    a3 += __shfl_xor(a3, off, 32);
    a4 += __shfl_xor(a4, off, 32);
    a5 += __shfl_xor(a5, off, 32);
  }
  if (s == 0 && eg < E) {
    const float bias = b2[0];
    const int node = edge_index[eg];
    float* ns = node_sum + (size_t)node * 9;
    const float r9[9] = {a0 + bias, a1 + bias, a2 + bias,
                         a1 + bias, a3 + bias, a4 + bias,
                         a2 + bias, a4 + bias, a5 + bias};
    #pragma unroll
    for (int k = 0; k < 9; ++k) atomicAdd(&ns[k], r9[k]);
    atomicAdd(&node_cnt[node], 1.0f);
  }
}

// --- node mean -> graph accumulate ------------------------------------------
__global__ void node_to_graph(const float* __restrict__ node_sum,
                              const float* __restrict__ node_cnt,
                              const int* __restrict__ batch,
                              float* __restrict__ gsum,
                              float* __restrict__ gcnt, int N) {
  int n = blockIdx.x * blockDim.x + threadIdx.x;
  if (n >= N) return;
  const float inv = 1.0f / fmaxf(node_cnt[n], 1.0f);
  const int g = batch[n];
  #pragma unroll
  for (int k = 0; k < 9; ++k)
    atomicAdd(&gsum[g * 9 + k], node_sum[(size_t)n * 9 + k] * inv);
  atomicAdd(&gcnt[g], 1.0f);
}

__global__ void finalize(const float* __restrict__ gsum,
                         const float* __restrict__ gcnt,
                         float* __restrict__ out, int G) {
  int i = blockIdx.x * blockDim.x + threadIdx.x;
  if (i >= G * 9) return;
  out[i] = gsum[i] / fmaxf(gcnt[i / 9], 1.0f);
}

extern "C" void kernel_launch(void* const* d_in, const int* in_sizes, int n_in,
                              void* d_out, int out_size, void* d_ws, size_t ws_size,
                              hipStream_t stream) {
  (void)n_in; (void)ws_size;
  const float* xv       = (const float*)d_in[0];
  const float* xedge    = (const float*)d_in[1];
  const float* W1       = (const float*)d_in[2];
  const float* b1       = (const float*)d_in[3];
  const float* W2       = (const float*)d_in[4];
  const float* b2       = (const float*)d_in[5];
  const int*   edge_idx = (const int*)d_in[6];
  const int*   batch    = (const int*)d_in[7];

  const int E = in_sizes[0] / 3;
  const int N = in_sizes[7];
  const int G = out_size / 9;

  float* node_sum = (float*)d_ws;                  // N*9
  float* node_cnt = node_sum + (size_t)N * 9;      // N
  float* gsum     = node_cnt + N;                  // G*9
  float* gcnt     = gsum + (size_t)G * 9;          // G
  uintptr_t p = (uintptr_t)(gcnt + G);
  p = (p + 255) & ~(uintptr_t)255;
  uint16_t* Whi = (uint16_t*)p;                    // 256*256 bf16
  uint16_t* Wlo = Whi + EMB * EMB;

  zero_f32<<<256, 256, 0, stream>>>(node_sum, N * 10 + G * 10);
  prep_w_split<<<(EMB * EMB + 255) / 256, 256, 0, stream>>>(W1, Whi, Wlo);
  edge_mlp_kernel<<<(E + TILE_E - 1) / TILE_E, 256, 0, stream>>>(
      xv, xedge, Whi, Wlo, b1, W2, b2, edge_idx, node_sum, node_cnt, E);
  node_to_graph<<<(N + 255) / 256, 256, 0, stream>>>(node_sum, node_cnt, batch,
                                                     gsum, gcnt, N);
  finalize<<<(G * 9 + 63) / 64, 64, 0, stream>>>(gsum, gcnt, (float*)d_out, G);
}
